// SimilarClusterEncoder_75522704933140
// MI455X (gfx1250) — matrix operand order repaired and checked by reading, not compiled, not run on hardware
//
#include <hip/hip_runtime.h>

// B=16, T=4096, D=256, K=2048, O=512
#define BT   65536    // B*T tokens
#define DD   256
#define KK   2048
#define OO   512

typedef __bf16 bf16_t;
typedef __attribute__((ext_vector_type(16))) __bf16 v16bf;
typedef __attribute__((ext_vector_type(8)))  float  v8f;

union FragA { bf16_t e[16]; v16bf v; };

// ---------------------------------------------------------------------------
// Kernel 1: centers -> bf16 hi/lo split + ||c||^2 per cluster.
// One block per cluster row, 256 threads over D.
// ---------------------------------------------------------------------------
__global__ __launch_bounds__(256)
void prep_centers_kernel(const float* __restrict__ C,
                         bf16_t* __restrict__ Chi,
                         bf16_t* __restrict__ Clo,
                         float* __restrict__ C2) {
    __shared__ float part[8];
    const int k = blockIdx.x;
    const int d = threadIdx.x;
    const float v = C[(size_t)k * DD + d];
    const bf16_t h = (bf16_t)v;
    Chi[(size_t)k * DD + d] = h;
    Clo[(size_t)k * DD + d] = (bf16_t)(v - (float)h);
    float s = v * v;
#pragma unroll
    for (int o = 16; o; o >>= 1) s += __shfl_down(s, o, 32);
    if ((threadIdx.x & 31) == 0) part[threadIdx.x >> 5] = s;
    __syncthreads();
    if (threadIdx.x == 0) {
        float t = 0.f;
#pragma unroll
        for (int i = 0; i < 8; ++i) t += part[i];
        C2[k] = t;
    }
}

// ---------------------------------------------------------------------------
// Kernel 2: proj[k][o] = centers[k] . W[o] + b[o]   (2048 x 512, fp32 VALU)
// One block per cluster; center row staged in LDS; each thread does 2 outputs.
// ---------------------------------------------------------------------------
__global__ __launch_bounds__(256)
void proj_kernel(const float* __restrict__ C,
                 const float* __restrict__ W,
                 const float* __restrict__ Bv,
                 float* __restrict__ P) {
    __shared__ float cs[DD];
    const int k = blockIdx.x;
    cs[threadIdx.x] = C[(size_t)k * DD + threadIdx.x];
    __syncthreads();
#pragma unroll
    for (int oo = 0; oo < 2; ++oo) {
        const int o = threadIdx.x + oo * 256;
        const float4* wr = (const float4*)(W + (size_t)o * DD);
        float acc = 0.f;
#pragma unroll 8
        for (int d4 = 0; d4 < DD / 4; ++d4) {
            const float4 w = wr[d4];
            const float* c4 = &cs[d4 * 4];
            acc = fmaf(w.x, c4[0], acc);
            acc = fmaf(w.y, c4[1], acc);
            acc = fmaf(w.z, c4[2], acc);
            acc = fmaf(w.w, c4[3], acc);
        }
        P[(size_t)k * OO + o] = acc + Bv[o];
    }
}

// ---------------------------------------------------------------------------
// Kernel 3: distance + argmin via split-bf16 WMMA.
// One wave = one 16-token M-tile; 8 waves/block. A (x tile, hi+lo) is
// converted once into registers; loop over 128 N-tiles of 16 clusters,
// 8 K-chunks of 32. The three split-products (hi*hi, hi*lo, lo*hi) go into
// THREE INDEPENDENT accumulator chains so back-to-back WMMAs never have a
// D->C RAW dependency (removes TRANS32_DEP delay slots); combined at the end.
// Score = c2[n] - 2 * dot  (||x||^2 is constant w.r.t. argmin).
// ---------------------------------------------------------------------------
__global__ __launch_bounds__(256)
void dist_argmin_kernel(const float* __restrict__ X,
                        const bf16_t* __restrict__ Chi,
                        const bf16_t* __restrict__ Clo,
                        const float* __restrict__ C2,
                        int* __restrict__ Sel) {
    const int lane = threadIdx.x & 31;
    const int wave = threadIdx.x >> 5;
    const int col  = lane & 15;   // N column (and A row) owned by this lane
    const int half = lane >> 4;   // K-half selector for A/B fragments
    const int m0   = (blockIdx.x * 8 + wave) * 16;

    // ---- Load + split the 16x256 x-tile into registers (A fragments) ----
    // Lane layout (contiguous k = 16*half + s, not AMD's table order): lanes<16 hold K {0..15} of row col;
    // lanes>=16 hold K {16..31}. Both runs are contiguous in memory.
    FragA ahi[8], alo[8];
    const float* xrow = X + (size_t)(m0 + col) * DD + half * 16;
#pragma unroll
    for (int kc = 0; kc < 8; ++kc) {
        const float* p = xrow + kc * 32;
#pragma unroll
        for (int i = 0; i < 8; ++i) {
            const float f0 = p[i];
            const float f1 = p[8 + i];
            const bf16_t h0 = (bf16_t)f0;
            const bf16_t h1 = (bf16_t)f1;
            ahi[kc].e[i]     = h0;
            alo[kc].e[i]     = (bf16_t)(f0 - (float)h0);
            ahi[kc].e[8 + i] = h1;
            alo[kc].e[8 + i] = (bf16_t)(f1 - (float)h1);
        }
    }

    float best[8];
    int   bidx[8];
#pragma unroll
    for (int e = 0; e < 8; ++e) { best[e] = 3.4e38f; bidx[e] = 0; }

    // ---- Sweep all 2048 clusters in N-tiles of 16 ----
    for (int n0 = 0; n0 < KK; n0 += 16) {
        // B layout (same contiguous k order as A): lane holds column n0+col, 16 contiguous
        // K values starting at half*16 -> a contiguous 32B run of the
        // row-major centers array. No transpose needed.
        const bf16_t* bh = Chi + (size_t)(n0 + col) * DD + half * 16;
        const bf16_t* bl = Clo + (size_t)(n0 + col) * DD + half * 16;

        if (n0 + 16 < KK) {  // gfx1250 global_prefetch of next tile
            __builtin_prefetch(Chi + (size_t)(n0 + 16 + col) * DD + half * 16, 0, 1);
            __builtin_prefetch(Clo + (size_t)(n0 + 16 + col) * DD + half * 16, 0, 1);
        }

        v8f acc_hh = {};   // hi(x) * hi(c)
        v8f acc_hl = {};   // hi(x) * lo(c)
        v8f acc_lh = {};   // lo(x) * hi(c)
#pragma unroll
        for (int kc = 0; kc < 8; ++kc) {
            const v16bf bhi = *(const v16bf*)(bh + kc * 32);
            const v16bf blo = *(const v16bf*)(bl + kc * 32);
            acc_hh = __builtin_amdgcn_wmma_f32_16x16x32_bf16(
                         false, ahi[kc].v, false, bhi, (short)0, acc_hh, false, false);
            acc_hl = __builtin_amdgcn_wmma_f32_16x16x32_bf16(
                         false, ahi[kc].v, false, blo, (short)0, acc_hl, false, false);
            acc_lh = __builtin_amdgcn_wmma_f32_16x16x32_bf16(
                         false, alo[kc].v, false, bhi, (short)0, acc_lh, false, false);
        }

        const float c2v = C2[n0 + col];
        const int   idx = n0 + col;
#pragma unroll
        for (int e = 0; e < 8; ++e) {
            const float dot = acc_hh[e] + acc_hl[e] + acc_lh[e];
            const float sc  = c2v - 2.0f * dot;
            if (sc < best[e]) { best[e] = sc; bidx[e] = idx; }  // strict <: first-index tie-break within lane
        }
    }

    // ---- Cross-lane min+argmin within each 16-lane half ----
    // best[e] corresponds to token m0 + half*8 + e, candidates spread over
    // the 16 lanes of this half (one N column per lane).
#pragma unroll
    for (int off = 8; off; off >>= 1) {
#pragma unroll
        for (int e = 0; e < 8; ++e) {
            const float ov = __shfl_xor(best[e], off, 32);
            const int   oi = __shfl_xor(bidx[e], off, 32);
            if (ov < best[e] || (ov == best[e] && oi < bidx[e])) {
                best[e] = ov; bidx[e] = oi;
            }
        }
    }
    if (col == 0) {  // lanes 0 and 16 hold finals for tokens m0+half*8+e
#pragma unroll
        for (int e = 0; e < 8; ++e) Sel[m0 + half * 8 + e] = bidx[e];
    }
}

// ---------------------------------------------------------------------------
// Kernel 4: out[m][:] = mask[m] ? proj[sel[m]][:] : 0   (bandwidth bound)
// 2 tokens per 256-thread block; one float4 per thread.
// ---------------------------------------------------------------------------
__global__ __launch_bounds__(256)
void gather_kernel(const float* __restrict__ P,
                   const int* __restrict__ Sel,
                   const unsigned char* __restrict__ Mask,
                   float* __restrict__ Out) {
    const int m = blockIdx.x * 2 + (threadIdx.x >> 7);
    const int j = threadIdx.x & 127;               // float4 index within row
    const int s = Sel[m];
    const float sc = Mask[m] ? 1.0f : 0.0f;        // np.bool_ = 1 byte
    float4 v = ((const float4*)(P + (size_t)s * OO))[j];
    v.x *= sc; v.y *= sc; v.z *= sc; v.w *= sc;
    ((float4*)(Out + (size_t)m * OO))[j] = v;
}

// ---------------------------------------------------------------------------
extern "C" void kernel_launch(void* const* d_in, const int* in_sizes, int n_in,
                              void* d_out, int out_size, void* d_ws, size_t ws_size,
                              hipStream_t stream) {
    const float*         x    = (const float*)d_in[0];          // [B,T,D]
    const float*         cc   = (const float*)d_in[1];          // [K,D]
    const float*         W    = (const float*)d_in[2];          // [O,D]
    const float*         bv   = (const float*)d_in[3];          // [O]
    const unsigned char* mask = (const unsigned char*)d_in[4];  // [B,T] bool
    float* out = (float*)d_out;                                  // [B,T,O]

    // Workspace carve-out (~6.6 MB total)
    char* ws = (char*)d_ws;
    bf16_t* c_hi = (bf16_t*)(ws);                               // 1 MB
    bf16_t* c_lo = (bf16_t*)(ws + (1u << 20));                  // 1 MB
    float*  c2   = (float*) (ws + (2u << 20));                  // 8 KB
    float*  proj = (float*) (ws + (2u << 20) + (1u << 15));     // 4 MB
    int*    sel  = (int*)   (ws + (6u << 20) + (1u << 15));     // 256 KB

    prep_centers_kernel<<<KK, 256, 0, stream>>>(cc, c_hi, c_lo, c2);
    proj_kernel<<<KK, 256, 0, stream>>>(cc, W, bv, proj);
    dist_argmin_kernel<<<BT / 128, 256, 0, stream>>>(x, c_hi, c_lo, c2, sel);
    gather_kernel<<<BT / 2, 256, 0, stream>>>(proj, sel, mask, out);
}
